// RGSGFMM_25821343384106
// MI455X (gfx1250) — matrix operand order repaired and checked by reading, not compiled, not run on hardware
//
#include <hip/hip_runtime.h>
#include <hip/hip_bf16.h>

typedef _Float16 half_t;
typedef _Float16 v16h __attribute__((ext_vector_type(16)));
typedef _Float16 v8h  __attribute__((ext_vector_type(8)));
typedef float    v8f  __attribute__((ext_vector_type(8)));

#define C_IN   128
#define H_DIM  32
#define NDEG   5          // degree+1 basis functions
#define ROWS   64         // rows per workgroup
#define KHALF  320        // K per half (64 channels * 5 degrees)
#define NKC    10         // 32-wide K chunks per half

// Native gfx1250 v_tanh_f32 if the toolchain exposes it; libm fallback else.
__device__ __forceinline__ float fast_tanh(float x) {
#if __has_builtin(__builtin_amdgcn_tanhf)
    return __builtin_amdgcn_tanhf(x);
#elif __has_builtin(__builtin_amdgcn_tanh_f32)
    return __builtin_amdgcn_tanh_f32(x);
#else
    return tanhf(x);
#endif
}

__device__ __forceinline__ float fast_rcp(float x) {
#if __has_builtin(__builtin_amdgcn_rcpf)
    return __builtin_amdgcn_rcpf(x);
#else
    return 1.f / x;
#endif
}

// ---------------------------------------------------------------------------
// Prep: repack coeffs1 [C][H][5] (f32) into WMMA B-fragment order (f16).
// ws element index: idx = (((half*10 + kc)*2 + ct)*32 + lane)*16 + i
// lane 0-15 -> klocal = 0..7,16..23; lane 16-31 -> klocal = 8..15,24..31 (i<8: 8*hi+i, i>=8: 16+8*hi+i-8)
// K map inside a half: k = d*64 + c_local,  c = half*64 + c_local
// ---------------------------------------------------------------------------
__global__ void RGSGFMM_repackB(const float* __restrict__ coeffs1,
                                half_t* __restrict__ Bws) {
    int idx = blockIdx.x * 256 + threadIdx.x;            // 0 .. 20479
    if (idx >= 2 * NKC * 2 * 32 * 16) return;
    int i    = idx & 15;
    int lane = (idx >> 4) & 31;
    int ct   = (idx >> 9) & 1;
    int kcg  = idx >> 10;                                // 0..19
    int kc   = kcg % NKC;
    int hf   = kcg / NKC;
    int klocal = ((lane >> 4) << 3) + (i & 7) + ((i >> 3) << 4); // 0..31
    int k_half = kc * 32 + klocal;                       // 0..319
    int d  = k_half >> 6;                                // /64 -> degree
    int cl = k_half & 63;
    int c  = hf * 64 + cl;
    int h  = ct * 16 + (lane & 15);
    Bws[idx] = (half_t)coeffs1[(c * H_DIM + h) * NDEG + d];
}

// ---------------------------------------------------------------------------
// Fused: ChebyKAN GEMM (WMMA f16->f32) -> LN -> ReLU -> ChebyKAN gate ->
//        sigmoid -> PRV * (1 + alpha*attn)
// ---------------------------------------------------------------------------
__global__ void __launch_bounds__(256)
RGSGFMM_fused(const float* __restrict__ PRV,
              const float* __restrict__ PR,
              const float* __restrict__ gamma,
              const float* __restrict__ beta,
              const float* __restrict__ coeffs2,
              const float* __restrict__ alphaPtr,
              const half_t* __restrict__ Bws,
              float* __restrict__ out) {
    __shared__ half_t sBasis[ROWS * KHALF];   // 40960 B
    __shared__ float  sCtx[ROWS * H_DIM];     //  8192 B
    __shared__ float  sAttn[ROWS];            //   256 B

    const int tid  = threadIdx.x;
    const int lane = tid & 31;
    const int wave = tid >> 5;
    const int rt   = wave >> 1;       // row tile 0..3
    const int ct   = wave & 1;        // col tile 0..1
    const long rowBase = (long)blockIdx.x * ROWS;

    v8f acc = {};

    for (int hf = 0; hf < 2; ++hf) {
        // ---- phase 1: Chebyshev basis of tanh(PR) into LDS (this K-half) ----
        // 64 rows * 32 channel-pairs = 2048 tasks
        #pragma unroll
        for (int it = 0; it < 8; ++it) {
            int task = it * 256 + tid;
            int r  = task >> 5;                 // row in block
            int jp = task & 31;                 // channel pair
            int c0 = hf * 64 + jp * 2;
            const float2 x2 = *(const float2*)(PR + (rowBase + r) * C_IN + c0);
            float t0 = fast_tanh(x2.x), t1 = fast_tanh(x2.y);
            float a0 = 1.f, a1 = t0, b0 = 1.f, b1 = t1;
            uint* dst = (uint*)(sBasis + r * KHALF + jp * 2);
            union { uint u; half_t h[2]; } pk;
            pk.h[0] = (half_t)a0; pk.h[1] = (half_t)b0; dst[0]      = pk.u; // d=0
            pk.h[0] = (half_t)a1; pk.h[1] = (half_t)b1; dst[32]     = pk.u; // d=1 (+64 halves)
            #pragma unroll
            for (int d = 2; d < NDEG; ++d) {
                float a2 = 2.f * t0 * a1 - a0;
                float b2 = 2.f * t1 * b1 - b0;
                pk.h[0] = (half_t)a2; pk.h[1] = (half_t)b2;
                dst[d * 32] = pk.u;
                a0 = a1; a1 = a2; b0 = b1; b1 = b2;
            }
        }
        __syncthreads();

        // ---- phase 2: WMMA over this K-half (uniform control flow) ----
        const int m  = lane & 15;
        const int hi = lane >> 4;
        #pragma unroll
        for (int kc = 0; kc < NKC; ++kc) {
            // A fragment: 16-bit 16x32 layout; lane-lo: K = kb+0..7 & kb+16..23,
            // lane-hi: K = kb+8..15 & kb+24..31
            const half_t* ap = sBasis + (rt * 16 + m) * KHALF + kc * 32 + hi * 8;
            v8h alo = *(const v8h*)ap;
            v8h ahi = *(const v8h*)(ap + 16);
            v16h A;
            #pragma unroll
            for (int i = 0; i < 8; ++i) { A[i] = alo[i]; A[i + 8] = ahi[i]; }

            // B fragment: prepacked, fully coalesced
            const half_t* bp = Bws + ((long)(((hf * NKC + kc) * 2 + ct) * 32 + lane)) * 16;
            v8h blo = *(const v8h*)bp;
            v8h bhi = *(const v8h*)(bp + 8);
            v16h B;
            #pragma unroll
            for (int i = 0; i < 8; ++i) { B[i] = blo[i]; B[i + 8] = bhi[i]; }

            acc = __builtin_amdgcn_wmma_f32_16x16x32_f16(
                      false, A, false, B, (short)0, acc, false, false);
        }
        __syncthreads();   // protect sBasis before next half overwrites it
    }

    // ---- scatter context (C/D layout: VGPR g -> M=g / M=g+8 per lane half) ----
    {
        const int m  = lane & 15;
        const int hi = lane >> 4;
        #pragma unroll
        for (int g = 0; g < 8; ++g) {
            int row = rt * 16 + hi * 8 + g;
            sCtx[row * H_DIM + ct * 16 + m] = acc[g];
        }
    }
    __syncthreads();

    // ---- phase 3: LayerNorm + ReLU + ChebyKAN gate + sigmoid (1 thread/row) --
    if (tid < ROWS) {
        float v[H_DIM];
        float mu = 0.f;
        #pragma unroll
        for (int h = 0; h < H_DIM; ++h) { v[h] = sCtx[tid * H_DIM + h]; mu += v[h]; }
        mu *= (1.f / H_DIM);
        float var = 0.f;
        #pragma unroll
        for (int h = 0; h < H_DIM; ++h) { float d = v[h] - mu; var += d * d; }
        var *= (1.f / H_DIM);
        float rstd = rsqrtf(var + 1e-5f);
        float s = 0.f;
        #pragma unroll
        for (int h = 0; h < H_DIM; ++h) {
            float y = (v[h] - mu) * rstd * gamma[h] + beta[h];
            y = fmaxf(y, 0.f);
            float t  = fast_tanh(y);
            float T2 = 2.f * t * t - 1.f;
            float T3 = 2.f * t * T2 - t;
            float T4 = 2.f * t * T3 - T2;
            const float* c2 = coeffs2 + h * NDEG;
            s += c2[0] + c2[1] * t + c2[2] * T2 + c2[3] * T3 + c2[4] * T4;
        }
        sAttn[tid] = fast_rcp(1.f + __expf(-s));
    }
    __syncthreads();

    // ---- phase 4: out = PRV * (1 + alpha*attn), float4 coalesced ----
    const float alpha = *alphaPtr;
    #pragma unroll
    for (int it = 0; it < 8; ++it) {
        int e  = it * 256 + tid;        // 2048 float4 per block
        int r  = e >> 5;                // 32 float4 per row
        int c4 = e & 31;
        long off = (rowBase + r) * C_IN + c4 * 4;
        float4 x = *(const float4*)(PRV + off);
        float sc = 1.f + alpha * sAttn[r];
        x.x *= sc; x.y *= sc; x.z *= sc; x.w *= sc;
        *(float4*)(out + off) = x;
    }
}

// ---------------------------------------------------------------------------
extern "C" void kernel_launch(void* const* d_in, const int* in_sizes, int n_in,
                              void* d_out, int out_size, void* d_ws, size_t ws_size,
                              hipStream_t stream) {
    const float* PRV    = (const float*)d_in[0];
    const float* PR     = (const float*)d_in[1];
    const float* c1     = (const float*)d_in[2];
    const float* gammap = (const float*)d_in[3];
    const float* betap  = (const float*)d_in[4];
    const float* c2     = (const float*)d_in[5];
    const float* alphap = (const float*)d_in[6];
    float*  out = (float*)d_out;
    half_t* Bws = (half_t*)d_ws;       // 20480 * 2 B = 40 KB scratch

    const int N = in_sizes[0] / C_IN;  // 262144

    RGSGFMM_repackB<<<80, 256, 0, stream>>>(c1, Bws);
    RGSGFMM_fused<<<N / ROWS, 256, 0, stream>>>(PRV, PR, gammap, betap, c2,
                                                alphap, Bws, out);
}
